// SparseAttention_48215302865704
// MI455X (gfx1250) — matrix operand order repaired and checked by reading, not compiled, not run on hardware
//
#include <hip/hip_runtime.h>
#include <hip/hip_bf16.h>

// ---------------------------------------------------------------------------
// BigBird-style sparse attention for MI455X (gfx1250).
// bf16 WMMA compute + CDNA5 async global->LDS staging (ASYNCcnt pipelined).
// ---------------------------------------------------------------------------

typedef __bf16 bf16;
typedef __attribute__((ext_vector_type(16))) __bf16 v16bf;
typedef __attribute__((ext_vector_type(8)))  float  v8f;

#define WMMA_BF16(a, b, c) \
  __builtin_amdgcn_wmma_f32_16x16x32_bf16(false, (a), false, (b), (short)0, (c), false, false)

constexpr int B_  = 2;
constexpr int S_  = 4096;
constexpr int E_  = 1024;
constexpr int H_  = 16;
constexpr int DH  = 64;
constexpr int GQ  = 64;           // global tokens
constexpr int BS  = 64;           // block size
constexpr int NB  = S_ / BS;      // 64 key blocks
constexpr int MB  = 12;           // sparse blocks per query block row
constexpr int NQB = NB - 1;       // 63 non-global query blocks

__device__ __forceinline__ v8f zero8() {
  v8f z;
#pragma unroll
  for (int i = 0; i < 8; ++i) z[i] = 0.0f;
  return z;
}

__device__ __forceinline__ float redmax16(float v) {
#pragma unroll
  for (int m = 1; m < 16; m <<= 1) v = fmaxf(v, __shfl_xor(v, m, 32));
  return v;
}
__device__ __forceinline__ float redsum16(float v) {
#pragma unroll
  for (int m = 1; m < 16; m <<= 1) v += __shfl_xor(v, m, 32);
  return v;
}

// ---- CDNA5 async global->LDS copy (16B per lane), tracked by ASYNCcnt ------
__device__ __forceinline__ void async_copy16(void* lds_ptr, const void* gptr) {
  const unsigned lds_addr = (unsigned)(unsigned long long)lds_ptr;  // ADDR[31:0] = LDS offset
  asm volatile("global_load_async_to_lds_b128 %0, %1, off"
               :: "v"(lds_addr), "v"((unsigned long long)gptr)
               : "memory");
}
__device__ __forceinline__ void wait_async0() {
  asm volatile("s_wait_asynccnt 0x0" ::: "memory");
}

// ---------------------------------------------------------------------------
// fp32 -> bf16 cast
// ---------------------------------------------------------------------------
__global__ void cast_bf16_kernel(const float* __restrict__ src,
                                 bf16* __restrict__ dst, int n) {
  int i = blockIdx.x * blockDim.x + threadIdx.x;
  int stride = gridDim.x * blockDim.x;
  for (; i < n; i += stride) dst[i] = (bf16)src[i];
}

// ---------------------------------------------------------------------------
// Tiled bf16 GEMM:  C[M,N] = A[M,K] * B[K,N]
// Block tile 128x128, 8 waves (4x2 wave grid), each wave 32x64, K-step 32.
// Double-buffered async global->LDS staging.
// ---------------------------------------------------------------------------
template <bool OUT_F32>
__global__ void gemm_bf16_kernel(const bf16* __restrict__ A,
                                 const bf16* __restrict__ Bm,
                                 void* __restrict__ Cout,
                                 int M, int N, int K) {
  __shared__ bf16 As[2][128 * 32];
  __shared__ bf16 Bs[2][32 * 128];

  const int tid  = threadIdx.x;          // 256 threads
  const int lane = tid & 31;
  const int wid  = tid >> 5;
  const int wm   = wid & 3;              // 4 M-strips of 32
  const int wn   = wid >> 2;             // 2 N-strips of 64
  const int hl   = lane >> 4;            // lane half
  const int ln   = lane & 15;

  const int rowBase = blockIdx.x * 128;
  const int colBase = blockIdx.y * 128;

  auto issue_tiles = [&](int kk, int buf) {
#pragma unroll
    for (int p = 0; p < 2; ++p) {
      const int lin = p * 2048 + tid * 8;
      const int ra = lin >> 5, ca = lin & 31;
      async_copy16(&As[buf][ra * 32 + ca],
                   &A[(size_t)(rowBase + ra) * K + kk + ca]);
      const int rb = lin >> 7, cb = lin & 127;
      async_copy16(&Bs[buf][rb * 128 + cb],
                   &Bm[(size_t)(kk + rb) * N + colBase + cb]);
    }
  };

  v8f acc[2][4];
#pragma unroll
  for (int mi = 0; mi < 2; ++mi)
#pragma unroll
    for (int ni = 0; ni < 4; ++ni) acc[mi][ni] = zero8();

  issue_tiles(0, 0);

  for (int kk = 0; kk < K; kk += 32) {
    const int cur = (kk >> 5) & 1;
    wait_async0();
    __syncthreads();
    if (kk + 32 < K) issue_tiles(kk + 32, cur ^ 1);

    // ---- build fragments (contiguous k = hl*16 + j in both A and B)
    v16bf af[2], bfr[4];
#pragma unroll
    for (int mi = 0; mi < 2; ++mi) {
      const int row = wm * 32 + mi * 16 + ln;
#pragma unroll
      for (int j = 0; j < 16; ++j) {
        const int kd = hl * 16 + j;
        af[mi][j] = As[cur][row * 32 + kd];
      }
    }
#pragma unroll
    for (int ni = 0; ni < 4; ++ni) {
      const int col = wn * 64 + ni * 16 + ln;
#pragma unroll
      for (int j = 0; j < 16; ++j) {
        const int kd = hl * 16 + j;
        bfr[ni][j] = Bs[cur][kd * 128 + col];
      }
    }
#pragma unroll
    for (int mi = 0; mi < 2; ++mi)
#pragma unroll
      for (int ni = 0; ni < 4; ++ni)
        acc[mi][ni] = WMMA_BF16(af[mi], bfr[ni], acc[mi][ni]);
  }

  // ---- epilogue
#pragma unroll
  for (int mi = 0; mi < 2; ++mi)
#pragma unroll
    for (int ni = 0; ni < 4; ++ni)
#pragma unroll
      for (int i = 0; i < 8; ++i) {
        const int row = rowBase + wm * 32 + mi * 16 + i + 8 * hl;
        const int col = colBase + wn * 64 + ni * 16 + ln;
        if (OUT_F32)
          ((float*)Cout)[(size_t)row * N + col] = acc[mi][ni][i];
        else
          ((bf16*)Cout)[(size_t)row * N + col] = (bf16)acc[mi][ni][i];
      }
}

// ---------------------------------------------------------------------------
// Flash-style attention over 64-key groups, double-buffered async K/V tiles.
// mode 0: global q rows [0,64), keys = all 64 groups.
// mode 1: sparse q block qb (rows 64+qb*64 ..), keys = global group + blocks.
// Q/K/V/out layout: [B][S][H*64] bf16 (head rows 128B-contiguous).
// ---------------------------------------------------------------------------
__global__ void attn_kernel(const bf16* __restrict__ Qb,
                            const bf16* __restrict__ Kb,
                            const bf16* __restrict__ Vb,
                            bf16* __restrict__ Ob,
                            const int* __restrict__ bidx, int mode) {
  __shared__ bf16 Qs[64 * 64];
  __shared__ bf16 Ks[2][64 * 64];
  __shared__ bf16 Vs[2][64 * 64];
  __shared__ bf16 Ps[4][16 * 32];
  __shared__ int  slist[16];             // valid key-group bases (mode 1)

  const int tid  = threadIdx.x;          // 128 threads = 4 waves
  const int lane = tid & 31;
  const int wave = tid >> 5;
  const int hl   = lane >> 4;
  const int ln   = lane & 15;

  int b, h, qb, q0;
  if (mode == 0) {
    b = blockIdx.x >> 4; h = blockIdx.x & 15; qb = -1; q0 = 0;
  } else {
    int id = blockIdx.x;
    qb = id % NQB; id /= NQB;
    h  = id % H_;
    b  = id / H_;
    q0 = GQ + qb * BS;
  }
  const float slope = exp2f(-0.5f * (float)(h + 1));  // alibi slope, H=16
  const float scale = 0.125f;                          // 1/sqrt(64)
  const size_t headoff = (size_t)h * DH;

  // ---- valid key-group list (uniform across block)
  int ngroups;
  if (mode == 0) {
    ngroups = NB;
  } else {
    if (tid == 0) {
      int n = 1;
      slist[0] = 0;                                    // global tokens first
      for (int j = 0; j < MB; ++j) {
        const int bb = bidx[(qb + 1) * MB + j];
        if (bb >= 0) slist[n++] = bb * BS;             // -1 block == masked out
      }
      slist[15] = n;
    }
    __syncthreads();
    ngroups = slist[15];
  }
  auto ks_of = [&](int gi) { return (mode == 0) ? gi * BS : slist[gi]; };

  auto issue_kv = [&](int ks0, int buf) {
#pragma unroll
    for (int p = 0; p < 4; ++p) {
      const int lin = p * 1024 + tid * 8;
      const int r = lin >> 6, c = lin & 63;
      const size_t src = ((size_t)b * S_ + ks0 + r) * E_ + headoff + c;
      async_copy16(&Ks[buf][r * 64 + c], &Kb[src]);
      async_copy16(&Vs[buf][r * 64 + c], &Vb[src]);
    }
  };

  // ---- stage Q block (64x64) + first K/V group asynchronously
#pragma unroll
  for (int p = 0; p < 4; ++p) {
    const int lin = p * 1024 + tid * 8;
    const int r = lin >> 6, c = lin & 63;
    async_copy16(&Qs[r * 64 + c],
                 &Qb[((size_t)b * S_ + q0 + r) * E_ + headoff + c]);
  }
  issue_kv(ks_of(0), 0);

  float rmax[8], rsum[8];
#pragma unroll
  for (int i = 0; i < 8; ++i) { rmax[i] = -3.0e30f; rsum[i] = 0.0f; }
  v8f acc[4];
#pragma unroll
  for (int c = 0; c < 4; ++c) acc[c] = zero8();

  for (int gi = 0; gi < ngroups; ++gi) {
    const int cur = gi & 1;
    const int ks0 = ks_of(gi);
    wait_async0();
    __syncthreads();
    if (gi + 1 < ngroups) issue_kv(ks_of(gi + 1), cur ^ 1);

#pragma unroll
    for (int st = 0; st < 2; ++st) {             // two 32-key steps per group
      const int kb32 = st * 32;

      // ---- scores: S = Q * K^T (two 16-col tiles, Dh=64 -> 2 K-chunks)
      v8f s0 = zero8(), s1 = zero8();
#pragma unroll
      for (int kc = 0; kc < 2; ++kc) {
        v16bf aq, k0, k1;
#pragma unroll
        for (int j = 0; j < 16; ++j) {
          const int kd = hl * 16 + j;
          aq[j] = Qs[(wave * 16 + ln) * 64 + kc * 32 + kd];
        }
#pragma unroll
        for (int j = 0; j < 16; ++j) {
          const int kd = kc * 32 + hl * 16 + j;  // contraction dim = head dim
          k0[j] = Ks[cur][(kb32 + ln) * 64 + kd];
          k1[j] = Ks[cur][(kb32 + 16 + ln) * 64 + kd];
        }
        s0 = WMMA_BF16(aq, k0, s0);
        s1 = WMMA_BF16(aq, k1, s1);
      }

      // ---- scale + alibi bias
      const int qpb = q0 + wave * 16 + 8 * hl;
      const float kp0 = (float)(ks0 + kb32 + ln);
#pragma unroll
      for (int i = 0; i < 8; ++i) {
        const float qp = (float)(qpb + i);
        s0[i] = s0[i] * scale - slope * fabsf(qp - kp0);
        s1[i] = s1[i] * scale - slope * fabsf(qp - (kp0 + 16.0f));
      }

      // ---- online softmax (row stats across 16-lane halves)
      float corr[8];
#pragma unroll
      for (int i = 0; i < 8; ++i) {
        float tm = redmax16(fmaxf(s0[i], s1[i]));
        const float nm = fmaxf(rmax[i], tm);
        const float c0 = __expf(rmax[i] - nm);
        const float p0 = __expf(s0[i] - nm);
        const float p1 = __expf(s1[i] - nm);
        rsum[i] = rsum[i] * c0 + redsum16(p0 + p1);
        rmax[i] = nm;
        corr[i] = c0;
        s0[i] = p0; s1[i] = p1;
      }
#pragma unroll
      for (int c = 0; c < 4; ++c)
#pragma unroll
        for (int i = 0; i < 8; ++i) acc[c][i] = acc[c][i] * corr[i];

      // ---- relayout P (C-frag -> A-frag) through per-wave LDS
      bf16* P = &Ps[wave][0];
#pragma unroll
      for (int i = 0; i < 8; ++i) {
        const int row = i + 8 * hl;
        P[row * 32 + ln]      = (bf16)s0[i];
        P[row * 32 + 16 + ln] = (bf16)s1[i];
      }
      __builtin_amdgcn_wave_barrier();           // keep DS store->load order

      v16bf pa;
#pragma unroll
      for (int j = 0; j < 16; ++j) {
        const int kd = hl * 16 + j;
        pa[j] = P[ln * 32 + kd];
      }
#pragma unroll
      for (int c = 0; c < 4; ++c) {
        v16bf bv;
#pragma unroll
        for (int j = 0; j < 16; ++j) {
          const int kd = hl * 16 + j;            // contraction dim = key
          bv[j] = Vs[cur][(kb32 + kd) * 64 + c * 16 + ln];
        }
        acc[c] = WMMA_BF16(pa, bv, acc[c]);
      }
    }
  }

  // ---- normalize and store (bf16, feeds the Wo GEMM)
#pragma unroll
  for (int c = 0; c < 4; ++c)
#pragma unroll
    for (int i = 0; i < 8; ++i) {
      const int row = q0 + wave * 16 + i + 8 * hl;
      const float o = acc[c][i] / rsum[i];
      Ob[((size_t)b * S_ + row) * E_ + headoff + c * 16 + ln] = (bf16)o;
    }
}

// ---------------------------------------------------------------------------
// Host launcher
// ---------------------------------------------------------------------------
extern "C" void kernel_launch(void* const* d_in, const int* in_sizes, int n_in,
                              void* d_out, int out_size, void* d_ws, size_t ws_size,
                              hipStream_t stream) {
  const float* x  = (const float*)d_in[0];
  const float* Wq = (const float*)d_in[1];
  const float* Wk = (const float*)d_in[2];
  const float* Wv = (const float*)d_in[3];
  const float* Wo = (const float*)d_in[4];
  const int*   bi = (const int*)d_in[5];
  float* out = (float*)d_out;

  const size_t nX = (size_t)B_ * S_ * E_;  // 8,388,608
  const size_t nW = (size_t)E_ * E_;       // 1,048,576

  char* w = (char*)d_ws;
  bf16* xb  = (bf16*)w; w += nX * sizeof(bf16);
  bf16* wqb = (bf16*)w; w += nW * sizeof(bf16);
  bf16* wkb = (bf16*)w; w += nW * sizeof(bf16);
  bf16* wvb = (bf16*)w; w += nW * sizeof(bf16);
  bf16* wob = (bf16*)w; w += nW * sizeof(bf16);
  bf16* qb  = (bf16*)w; w += nX * sizeof(bf16);
  bf16* kb  = (bf16*)w; w += nX * sizeof(bf16);
  bf16* vb  = (bf16*)w; w += nX * sizeof(bf16);
  bf16* ao  = (bf16*)w; w += nX * sizeof(bf16);

  // 1) cast inputs to bf16
  cast_bf16_kernel<<<4096, 256, 0, stream>>>(x,  xb,  (int)nX);
  cast_bf16_kernel<<<1024, 256, 0, stream>>>(Wq, wqb, (int)nW);
  cast_bf16_kernel<<<1024, 256, 0, stream>>>(Wk, wkb, (int)nW);
  cast_bf16_kernel<<<1024, 256, 0, stream>>>(Wv, wvb, (int)nW);
  cast_bf16_kernel<<<1024, 256, 0, stream>>>(Wo, wob, (int)nW);

  // 2) QKV projections (bf16 out, x-layout [B][S][H*64])
  const dim3 gg(64, 8, 1);  // (8192/128, 1024/128)
  gemm_bf16_kernel<false><<<gg, 256, 0, stream>>>(xb, wqb, qb, 8192, 1024, 1024);
  gemm_bf16_kernel<false><<<gg, 256, 0, stream>>>(xb, wkb, kb, 8192, 1024, 1024);
  gemm_bf16_kernel<false><<<gg, 256, 0, stream>>>(xb, wvb, vb, 8192, 1024, 1024);

  // 3) attention: global rows then sparse query blocks
  attn_kernel<<<B_ * H_,        128, 0, stream>>>(qb, kb, vb, ao, bi, 0);
  attn_kernel<<<B_ * H_ * NQB,  128, 0, stream>>>(qb, kb, vb, ao, bi, 1);

  // 4) output projection (fp32 to d_out)
  gemm_bf16_kernel<true><<<gg, 256, 0, stream>>>(ao, wob, out, 8192, 1024, 1024);
}
